// DecodeBlock_23021024706723
// MI455X (gfx1250) — matrix operand order repaired and checked by reading, not compiled, not run on hardware
//
#include <hip/hip_runtime.h>
#include <math.h>

// ---------------------------------------------------------------------------
// CDNA5 (gfx1250) decoder block: wave32, v_wmma_f32_16x16x32_bf16 core,
// vectorized fragment staging, double-buffered LDS, L2-friendly bias stream.
// ---------------------------------------------------------------------------

typedef __bf16 bf16;
typedef __attribute__((ext_vector_type(16))) __bf16 v16bf;
typedef __attribute__((ext_vector_type(8)))  __bf16 v8bf;
typedef __attribute__((ext_vector_type(8)))  float  v8f;

#define D_MODEL 1024
#define NHEADS  16
#define DHEAD   64

static __device__ __forceinline__ v8f wmma_bf16(v16bf a, v16bf b, v8f c) {
  // (neg_a, A, neg_b, B, c_mod, C, reuse_a, reuse_b)
  return __builtin_amdgcn_wmma_f32_16x16x32_bf16(false, a, false, b, (short)0, c,
                                                 false, false);
}

// ---- WMMA fragment layout helpers (ISA 7.12.2) ----------------------------
// A (16x32): lanes 0-15 hold K={0..7,16..23}, lanes 16-31 hold K={8..15,24..31}
static __device__ __forceinline__ int a_lane(int m, int k) {
  return (((k & 15) >= 8) ? 16 : 0) + m;
}
static __device__ __forceinline__ int a_elem(int k) {
  int half = ((k & 15) >= 8) ? 1 : 0;
  int kk = k - half * 8;
  int v = (kk < 8) ? (kk >> 1) : (4 + ((kk - 16) >> 1));
  return v * 2 + (kk & 1);
}
static __device__ __forceinline__ int a_k_of(int lane, int e) {
  int half = lane >> 4;
  int v = e >> 1, p = e & 1;
  return ((v < 4) ? (v * 2 + p) : (16 + (v - 4) * 2 + p)) + half * 8;
}
static __device__ __forceinline__ float gelu_exact(float x) {
  return 0.5f * x * (1.0f + erff(x * 0.70710678118654752f));
}

// ---------------------------------------------------------------------------
// GEMM: C[M,N] = act(A[M,K] @ W[K,N] + bias[N]).  fp32 in/out, bf16 WMMA.
// Block 256 thr = 8 waves; tile 128x128x32; wave = 64(M) x 32(N); dbl-buffered.
// ---------------------------------------------------------------------------
template <int ACT, int N, int K>
__global__ __launch_bounds__(256) void gemm_bias_kernel(
    const float* __restrict__ A, const float* __restrict__ W,
    const float* __restrict__ bias, float* __restrict__ C, int M) {
  __shared__ __align__(32) bf16 sA[2][8 * 512];   // 8 A-frags per buffer
  __shared__ __align__(32) bf16 sB[2][8 * 512];   // 8 B-frags per buffer
  const int tid  = threadIdx.x;
  const int wave = tid >> 5, lane = tid & 31;
  const int bm = blockIdx.y * 128, bn = blockIdx.x * 128;
  const int wmf = (wave >> 2) * 4;                // A-frag base (0 or 4)
  const int wnf = (wave & 3) * 2;                 // B-frag base (0,2,4,6)

  // Stage one 128x32 A tile + 32x128 W tile into fragment-layout LDS.
  auto stage = [&](int k0, int p) {
#pragma unroll
    for (int ch = 0; ch < 2; ++ch) {
      int cid = ch * 256 + tid;
      int r = cid >> 2, s = cid & 3;              // row, 8-wide k segment
      const float4* ap = (const float4*)(A + (size_t)(bm + r) * K + k0 + s * 8);
      float4 f0 = ap[0], f1 = ap[1];
      v8bf av;
      av[0] = (bf16)f0.x; av[1] = (bf16)f0.y; av[2] = (bf16)f0.z; av[3] = (bf16)f0.w;
      av[4] = (bf16)f1.x; av[5] = (bf16)f1.y; av[6] = (bf16)f1.z; av[7] = (bf16)f1.w;
      int lane_ = ((s >> 1) << 4) + (r & 15);     // k-group -> lane half
      *(v8bf*)&sA[p][(r >> 4) * 512 + lane_ * 16 + (s & 1) * 8] = av;
    }
    {
      int c = tid & 127, hf = tid >> 7;           // column, k half (16 rows)
      const float* wp = W + (size_t)(k0 + hf * 16) * N + bn + c;
      v16bf wv;
#pragma unroll
      for (int i = 0; i < 16; ++i) wv[i] = (bf16)wp[(size_t)i * N];
      *(v16bf*)&sB[p][(c >> 4) * 512 + ((hf << 4) + (c & 15)) * 16] = wv;
    }
  };

  v8f acc[4][2];
#pragma unroll
  for (int i = 0; i < 4; ++i)
#pragma unroll
    for (int j = 0; j < 2; ++j)
#pragma unroll
      for (int e = 0; e < 8; ++e) acc[i][j][e] = 0.0f;

  stage(0, 0);
  int p = 0;
  for (int k0 = 0; k0 < K; k0 += 32) {
    __syncthreads();
    if (k0 + 32 < K) stage(k0 + 32, p ^ 1);       // overlap with WMMA below
    v16bf af[4], bfr[2];
#pragma unroll
    for (int i = 0; i < 4; ++i)
      af[i] = *(const v16bf*)&sA[p][(wmf + i) * 512 + lane * 16];
#pragma unroll
    for (int j = 0; j < 2; ++j)
      bfr[j] = *(const v16bf*)&sB[p][(wnf + j) * 512 + lane * 16];
#pragma unroll
    for (int i = 0; i < 4; ++i)
#pragma unroll
      for (int j = 0; j < 2; ++j) acc[i][j] = wmma_bf16(af[i], bfr[j], acc[i][j]);
    p ^= 1;
  }

  const int mloc = (lane >> 4) * 8, ncol = lane & 15;
#pragma unroll
  for (int j = 0; j < 2; ++j) {
    int gn = bn + (wnf + j) * 16 + ncol;
    float bv = bias[gn];
#pragma unroll
    for (int i = 0; i < 4; ++i) {
#pragma unroll
      for (int e = 0; e < 8; ++e) {
        int gm = bm + (wmf + i) * 16 + mloc + e;
        float v = acc[i][j][e] + bv;
        if (ACT == 1) v = gelu_exact(v);
        C[(size_t)gm * N + gn] = v;
      }
    }
  }
}

// ---------------------------------------------------------------------------
// Flash attention, additive relative bias (added BEFORE 1/sqrt(dh)), key mask,
// optional causal.  Q,K,V: [B*Lx, 1024], head h at cols h*64..+63.
// Grid: (H, Lq/64, B) so all heads of one q-tile share rel_mat cachelines.
// Block 128 thr = 4 waves; wave owns 16 q rows; double-buffered K/V tiles.
// ---------------------------------------------------------------------------
template <int LQ, int LK, int CAUSAL>
__global__ __launch_bounds__(128) void attn_kernel(
    const float* __restrict__ Q, const float* __restrict__ Kc,
    const float* __restrict__ Vc, const float* __restrict__ bias,
    const unsigned char* __restrict__ kmask, float* __restrict__ O) {
  __shared__ __align__(32) bf16 sK[2][4 * 512];   // [dh-chunk][k-tile] B-frags
  __shared__ __align__(32) bf16 sV[2][4 * 512];   // [dh-tile] B-frags
  __shared__ __align__(32) bf16 sP[4 * 512];      // per-wave P as A-frag
  const int tid  = threadIdx.x;
  const int wave = tid >> 5, lane = tid & 31;
  const int h = blockIdx.x, qt = blockIdx.y, b = blockIdx.z;
  const int q0 = qt * 64 + wave * 16;
  const size_t qrow0 = (size_t)b * LQ + q0;
  const int mloc = (lane >> 4) * 8, ncol = lane & 15;

  // Stage 32x64 K and V tiles (fragment-layout, vectorized stores).
  auto stage = [&](int k0, int p) {
    {                                             // K: contraction = dh
      int kk = tid >> 2, s = tid & 3;             // k row, 16-wide d segment
      const float4* kp =
          (const float4*)(Kc + ((size_t)b * LK + k0 + kk) * D_MODEL + h * DHEAD + s * 16);
      float4 f0 = kp[0], f1 = kp[1], f2 = kp[2], f3 = kp[3];
      v16bf kv;
      kv[0]  = (bf16)f0.x; kv[1]  = (bf16)f0.y; kv[2]  = (bf16)f0.z; kv[3]  = (bf16)f0.w;
      kv[4]  = (bf16)f1.x; kv[5]  = (bf16)f1.y; kv[6]  = (bf16)f1.z; kv[7]  = (bf16)f1.w;
      kv[8]  = (bf16)f2.x; kv[9]  = (bf16)f2.y; kv[10] = (bf16)f2.z; kv[11] = (bf16)f2.w;
      kv[12] = (bf16)f3.x; kv[13] = (bf16)f3.y; kv[14] = (bf16)f3.z; kv[15] = (bf16)f3.w;
      int frag  = (s >> 1) * 2 + (kk >> 4);
      int lane_ = ((s & 1) << 4) + (kk & 15);
      *(v16bf*)&sK[p][frag * 512 + lane_ * 16] = kv;
    }
    {                                             // V: contraction = k
      int d = tid >> 1, ks = tid & 1;             // d column, 16-wide k segment
      const float* vp = Vc + ((size_t)b * LK + k0 + ks * 16) * D_MODEL + h * DHEAD + d;
      v16bf vv;
#pragma unroll
      for (int i = 0; i < 16; ++i) vv[i] = (bf16)vp[(size_t)i * D_MODEL];
      *(v16bf*)&sV[p][(d >> 4) * 512 + ((ks << 4) + (d & 15)) * 16] = vv;
    }
  };

  // Q fragments (A-layout) gathered straight into registers, held all loop.
  v16bf qa[2];
#pragma unroll
  for (int c = 0; c < 2; ++c) {
    const float* qp = Q + (qrow0 + (lane & 15)) * D_MODEL + h * DHEAD + c * 32;
#pragma unroll
    for (int e = 0; e < 16; ++e) qa[c][e] = (bf16)qp[(lane >> 4) * 16 + e];
  }

  v8f o[4];
  float mrow[8], lrow[8];
#pragma unroll
  for (int j = 0; j < 4; ++j)
#pragma unroll
    for (int e = 0; e < 8; ++e) o[j][e] = 0.0f;
#pragma unroll
  for (int e = 0; e < 8; ++e) { mrow[e] = -3.0e38f; lrow[e] = 0.0f; }

  const int kend = CAUSAL ? (qt * 64 + 64) : LK;
  stage(0, 0);
  int p = 0;
  for (int k0 = 0; k0 < kend; k0 += 32) {
    __syncthreads();
    if (k0 + 32 < kend) stage(k0 + 32, p ^ 1);
    if (!(CAUSAL && k0 > q0 + 15)) {              // wave fully above diagonal?
      // S = Q * K^T : two 16x16 tiles, 2 WMMAs each (dh = 2 x 32).
      v8f s[2];
#pragma unroll
      for (int tt = 0; tt < 2; ++tt) {
#pragma unroll
        for (int e = 0; e < 8; ++e) s[tt][e] = 0.0f;
        v16bf kb0 = *(const v16bf*)&sK[p][(0 * 2 + tt) * 512 + lane * 16];
        v16bf kb1 = *(const v16bf*)&sK[p][(1 * 2 + tt) * 512 + lane * 16];
        s[tt] = wmma_bf16(qa[0], kb0, s[tt]);
        s[tt] = wmma_bf16(qa[1], kb1, s[tt]);
      }

      // logits = (S + bias) * 1/sqrt(64), causal + key mask; prefetch next.
      float lg[2][8];
#pragma unroll
      for (int tt = 0; tt < 2; ++tt) {
        int kg = k0 + tt * 16 + ncol;
        bool kvalid = kmask[(size_t)b * LK + kg] != 0;
#pragma unroll
        for (int e = 0; e < 8; ++e) {
          int qg = q0 + mloc + e;
          const float* bp = &bias[(((size_t)b * LQ + qg) * LK + kg) * NHEADS + h];
          float bv = *bp;
          if (kg + 32 < kend) __builtin_prefetch(bp + 32 * NHEADS, 0, 1);
          float xv = (s[tt][e] + bv) * 0.125f;
          bool masked = (!kvalid) || (CAUSAL && kg > qg);
          lg[tt][e] = masked ? -1.0e30f : xv;
        }
      }

      // Online softmax: a row lives in one half-wave (C layout) -> xor reduce.
      float alpha[8];
#pragma unroll
      for (int e = 0; e < 8; ++e) {
        float mx = fmaxf(lg[0][e], lg[1][e]);
#pragma unroll
        for (int off = 1; off < 16; off <<= 1) mx = fmaxf(mx, __shfl_xor(mx, off, 32));
        float mnew = fmaxf(mrow[e], mx);
        alpha[e] = __expf(mrow[e] - mnew);
        float p0 = __expf(lg[0][e] - mnew);
        float p1 = __expf(lg[1][e] - mnew);
        lg[0][e] = p0; lg[1][e] = p1;
        float rsum = p0 + p1;
#pragma unroll
        for (int off = 1; off < 16; off <<= 1) rsum += __shfl_xor(rsum, off, 32);
        lrow[e] = lrow[e] * alpha[e] + rsum;
        mrow[e] = mnew;
      }
#pragma unroll
      for (int j = 0; j < 4; ++j)
#pragma unroll
        for (int e = 0; e < 8; ++e) o[j][e] *= alpha[e];

      // Restage P (C-layout -> A-fragment) via per-wave LDS region.
#pragma unroll
      for (int tt = 0; tt < 2; ++tt)
#pragma unroll
        for (int e = 0; e < 8; ++e) {
          int n = tt * 16 + ncol;
          sP[wave * 512 + ((n >> 4) * 16 + mloc + e) * 16 + (n & 15)] = (bf16)lg[tt][e];
        }
      asm volatile("s_wait_dscnt 0" ::: "memory");  // DS in-order per wave
      v16bf pa = *(const v16bf*)&sP[wave * 512 + lane * 16];
#pragma unroll
      for (int j = 0; j < 4; ++j) {
        v16bf vb = *(const v16bf*)&sV[p][j * 512 + lane * 16];
        o[j] = wmma_bf16(pa, vb, o[j]);
      }
    }
    p ^= 1;
  }

  // Normalize and write merged heads: O[b,q, h*64 + d].
#pragma unroll
  for (int e = 0; e < 8; ++e) {
    float inv = (lrow[e] > 0.0f) ? (1.0f / lrow[e]) : 0.0f;
#pragma unroll
    for (int j = 0; j < 4; ++j)
      O[(qrow0 + mloc + e) * D_MODEL + h * DHEAD + j * 16 + ncol] = o[j][e] * inv;
  }
}

// ---------------------------------------------------------------------------
// Out = LN(X + Y) * gam + bet     (block per row, D=1024, in-place safe)
// ---------------------------------------------------------------------------
__global__ __launch_bounds__(256) void add_ln_kernel(
    const float* __restrict__ X, const float* __restrict__ Y,
    const float* __restrict__ gam, const float* __restrict__ bet,
    float* __restrict__ Out) {
  const int row = blockIdx.x, tid = threadIdx.x;
  const size_t base = (size_t)row * D_MODEL;
  __shared__ float rs[8], rs2[8];
  float v[4], s = 0.0f, s2 = 0.0f;
#pragma unroll
  for (int i = 0; i < 4; ++i) {
    int c = tid + i * 256;
    float t = X[base + c] + Y[base + c];
    v[i] = t; s += t; s2 += t * t;
  }
#pragma unroll
  for (int off = 1; off < 32; off <<= 1) {
    s += __shfl_xor(s, off, 32); s2 += __shfl_xor(s2, off, 32);
  }
  if ((tid & 31) == 0) { rs[tid >> 5] = s; rs2[tid >> 5] = s2; }
  __syncthreads();
  float ts = 0.0f, ts2 = 0.0f;
#pragma unroll
  for (int i = 0; i < 8; ++i) { ts += rs[i]; ts2 += rs2[i]; }
  float mean = ts * (1.0f / D_MODEL);
  float var  = ts2 * (1.0f / D_MODEL) - mean * mean;
  float rstd = rsqrtf(var + 1e-5f);
#pragma unroll
  for (int i = 0; i < 4; ++i) {
    int c = tid + i * 256;
    Out[base + c] = (v[i] - mean) * rstd * gam[c] + bet[c];
  }
}

// g[row] = sigmoid(dot(X[row], gw) + gb)   (wave per row)
__global__ __launch_bounds__(256) void gate_kernel(
    const float* __restrict__ X, const float* __restrict__ gw,
    const float* __restrict__ gb, float* __restrict__ G, int Mrows) {
  int row = blockIdx.x * 8 + (threadIdx.x >> 5);
  int lane = threadIdx.x & 31;
  if (row >= Mrows) return;
  const float* x = X + (size_t)row * D_MODEL;
  float s = 0.0f;
#pragma unroll
  for (int i = 0; i < D_MODEL / 32; ++i) s += x[lane + i * 32] * gw[lane + i * 32];
#pragma unroll
  for (int off = 1; off < 32; off <<= 1) s += __shfl_xor(s, off, 32);
  if (lane == 0) G[row] = 1.0f / (1.0f + __expf(-(s + gb[0])));
}

// Out = LN(X + (1-g)*Ys + g*Yr) * gam + bet
__global__ __launch_bounds__(256) void combine_ln_kernel(
    const float* __restrict__ X, const float* __restrict__ Ys,
    const float* __restrict__ Yr, const float* __restrict__ G,
    const float* __restrict__ gam, const float* __restrict__ bet,
    float* __restrict__ Out) {
  const int row = blockIdx.x, tid = threadIdx.x;
  const size_t base = (size_t)row * D_MODEL;
  const float gg = G[row];
  __shared__ float rs[8], rs2[8];
  float v[4], s = 0.0f, s2 = 0.0f;
#pragma unroll
  for (int i = 0; i < 4; ++i) {
    int c = tid + i * 256;
    float t = X[base + c] + (1.0f - gg) * Ys[base + c] + gg * Yr[base + c];
    v[i] = t; s += t; s2 += t * t;
  }
#pragma unroll
  for (int off = 1; off < 32; off <<= 1) {
    s += __shfl_xor(s, off, 32); s2 += __shfl_xor(s2, off, 32);
  }
  if ((tid & 31) == 0) { rs[tid >> 5] = s; rs2[tid >> 5] = s2; }
  __syncthreads();
  float ts = 0.0f, ts2 = 0.0f;
#pragma unroll
  for (int i = 0; i < 8; ++i) { ts += rs[i]; ts2 += rs2[i]; }
  float mean = ts * (1.0f / D_MODEL);
  float var  = ts2 * (1.0f / D_MODEL) - mean * mean;
  float rstd = rsqrtf(var + 1e-5f);
#pragma unroll
  for (int i = 0; i < 4; ++i) {
    int c = tid + i * 256;
    Out[base + c] = (v[i] - mean) * rstd * gam[c] + bet[c];
  }
}

// ---------------------------------------------------------------------------
extern "C" void kernel_launch(void* const* d_in, const int* in_sizes, int n_in,
                              void* d_out, int out_size, void* d_ws, size_t ws_size,
                              hipStream_t stream) {
  (void)in_sizes; (void)n_in; (void)out_size; (void)ws_size;
  const float* x        = (const float*)d_in[0];
  const float* rel_mat  = (const float*)d_in[1];
  const float* enc_rel  = (const float*)d_in[2];
  const float* rep_enc  = (const float*)d_in[3];
  const unsigned char* dec_mask = (const unsigned char*)d_in[4];
  const unsigned char* enc_mask = (const unsigned char*)d_in[5];
  // d_in[6] = token_type (0 -> req_assign expert path, as in reference)
  const float* a1_wq = (const float*)d_in[7];  const float* a1_bq = (const float*)d_in[8];
  const float* a1_wk = (const float*)d_in[9];  const float* a1_bk = (const float*)d_in[10];
  const float* a1_wv = (const float*)d_in[11]; const float* a1_bv = (const float*)d_in[12];
  const float* a1_wo = (const float*)d_in[13]; const float* a1_bo = (const float*)d_in[14];
  const float* a2_wk = (const float*)d_in[15]; const float* a2_bk = (const float*)d_in[16];
  const float* a2_wo = (const float*)d_in[17]; const float* a2_bo = (const float*)d_in[18];
  const float* ln1_g = (const float*)d_in[19]; const float* ln1_b = (const float*)d_in[20];
  const float* ln2_g = (const float*)d_in[21]; const float* ln2_b = (const float*)d_in[22];
  const float* ln3_g = (const float*)d_in[23]; const float* ln3_b = (const float*)d_in[24];
  const float* sh_w1 = (const float*)d_in[25]; const float* sh_b1 = (const float*)d_in[26];
  const float* sh_w2 = (const float*)d_in[27]; const float* sh_b2 = (const float*)d_in[28];
  const float* ra_w1 = (const float*)d_in[29]; const float* ra_b1 = (const float*)d_in[30];
  const float* ra_w2 = (const float*)d_in[31]; const float* ra_b2 = (const float*)d_in[32];
  const float* gate_w = (const float*)d_in[33]; const float* gate_b = (const float*)d_in[34];

  const int B = 2, L = 2048, LE = 1024;
  const int Mq = B * L;    // 4096 decoder tokens
  const int Me = B * LE;   // 2048 encoder tokens

  // Workspace carve-out (floats), ~184 MB total.
  float* ws = (float*)d_ws;
  size_t off = 0;
  float* bufQ  = ws + off; off += (size_t)Mq * D_MODEL;
  float* bufK  = ws + off; off += (size_t)Mq * D_MODEL;
  float* bufV  = ws + off; off += (size_t)Mq * D_MODEL;
  float* bufA  = ws + off; off += (size_t)Mq * D_MODEL;       // attn out
  float* bufX  = ws + off; off += (size_t)Mq * D_MODEL;       // running x
  float* bufK2 = ws + off; off += (size_t)Me * D_MODEL;
  float* bufH  = ws + off; off += (size_t)Mq * 4 * D_MODEL;   // FFN hidden (reused)
  float* bufYs = ws + off; off += (size_t)Mq * D_MODEL;
  float* bufYr = ws + off; off += (size_t)Mq * D_MODEL;
  float* bufG  = ws + off; off += (size_t)Mq;

  const dim3 blk256(256);
  const dim3 gProj(D_MODEL / 128, Mq / 128);          // 8 x 32
  const dim3 gProjE(D_MODEL / 128, Me / 128);         // 8 x 16
  const dim3 gFfn1(4 * D_MODEL / 128, Mq / 128);      // 32 x 32
  const dim3 gAttn(NHEADS, L / 64, B);                // heads adjacent -> L2 reuse
  const dim3 blkAttn(128);

  // ---- self attention ----
  gemm_bias_kernel<0, 1024, 1024><<<gProj, blk256, 0, stream>>>(x, a1_wq, a1_bq, bufQ, Mq);
  gemm_bias_kernel<0, 1024, 1024><<<gProj, blk256, 0, stream>>>(x, a1_wk, a1_bk, bufK, Mq);
  gemm_bias_kernel<0, 1024, 1024><<<gProj, blk256, 0, stream>>>(x, a1_wv, a1_bv, bufV, Mq);
  attn_kernel<2048, 2048, 1><<<gAttn, blkAttn, 0, stream>>>(bufQ, bufK, bufV, rel_mat,
                                                            dec_mask, bufA);
  gemm_bias_kernel<0, 1024, 1024><<<gProj, blk256, 0, stream>>>(bufA, a1_wo, a1_bo, bufQ, Mq);
  add_ln_kernel<<<dim3(Mq), blk256, 0, stream>>>(x, bufQ, ln1_g, ln1_b, bufX);

  // ---- cross attention (q/v identity, key projected) ----
  gemm_bias_kernel<0, 1024, 1024><<<gProjE, blk256, 0, stream>>>(rep_enc, a2_wk, a2_bk, bufK2, Me);
  attn_kernel<2048, 1024, 0><<<gAttn, blkAttn, 0, stream>>>(bufX, bufK2, rep_enc, enc_rel,
                                                            enc_mask, bufA);
  gemm_bias_kernel<0, 1024, 1024><<<gProj, blk256, 0, stream>>>(bufA, a2_wo, a2_bo, bufQ, Mq);
  add_ln_kernel<<<dim3(Mq), blk256, 0, stream>>>(bufX, bufQ, ln2_g, ln2_b, bufX);

  // ---- gated MoE FFN ----
  gemm_bias_kernel<1, 4096, 1024><<<gFfn1, blk256, 0, stream>>>(bufX, sh_w1, sh_b1, bufH, Mq);
  gemm_bias_kernel<0, 1024, 4096><<<gProj, blk256, 0, stream>>>(bufH, sh_w2, sh_b2, bufYs, Mq);
  gemm_bias_kernel<1, 4096, 1024><<<gFfn1, blk256, 0, stream>>>(bufX, ra_w1, ra_b1, bufH, Mq);
  gemm_bias_kernel<0, 1024, 4096><<<gProj, blk256, 0, stream>>>(bufH, ra_w2, ra_b2, bufYr, Mq);
  gate_kernel<<<dim3(Mq / 8), blk256, 0, stream>>>(bufX, gate_w, gate_b, bufG, Mq);
  combine_ln_kernel<<<dim3(Mq), blk256, 0, stream>>>(bufX, bufYs, bufYr, bufG, ln3_g, ln3_b,
                                                     (float*)d_out);
}
